// CESFDA_57956288692798
// MI455X (gfx1250) — matrix operand order repaired and checked by reading, not compiled, not run on hardware
//
#include <hip/hip_runtime.h>
#include <hip/hip_bf16.h>
#include <math.h>

typedef __attribute__((ext_vector_type(16))) _Float16 v16h;
typedef __attribute__((ext_vector_type(8)))  float    v8f;

#define B_      256
#define D_      512
#define C_      10
#define N_      200000
#define CHUNK   512
#define NCHUNK  ((N_ + CHUNK - 1) / CHUNK)      /* 391 */

/* workspace layout (bytes) */
#define OFF_FN16  0                               /* 256*512 f16  = 262144 */
#define OFF_FN32  262144                          /* 256*512 f32  = 524288 */
#define OFF_ZERO  786432                          /* 512 f32      =   2048 */
#define OFF_PAAD  788480                          /* 256*10 f32   =  10240 */
#define OFF_OWNER 798720                          /* 200000 i32   = 800000 */
#define OFF_ACCUM 1598720                         /* 16 f32: ent,ce,kl,mp[10] */
#define OFF_CANDV 1598784                         /* 256*391*6 f32 */
#define CAND_ELEMS (B_ * NCHUNK * 6)
#define OFF_CANDI (OFF_CANDV + CAND_ELEMS * 4)

#define TOPK_INSERT(tv, ti, V, J)                                          \
  do {                                                                     \
    float nv_ = (V); int nj_ = (J);                                        \
    _Pragma("unroll")                                                      \
    for (int s_ = 0; s_ < 6; ++s_) {                                       \
      bool gt_ = nv_ > tv[s_];                                             \
      float cv_ = tv[s_]; int cj_ = ti[s_];                                \
      tv[s_] = gt_ ? nv_ : cv_; ti[s_] = gt_ ? nj_ : cj_;                  \
      nv_ = gt_ ? cv_ : nv_;    nj_ = gt_ ? cj_ : nj_;                     \
    }                                                                      \
  } while (0)

/* ---------------- init: owner map + accumulators + zero row ---------------- */
__global__ void k_init(int* owner, float* accum, float* zerorow) {
  int i = blockIdx.x * blockDim.x + threadIdx.x;
  if (i < N_) owner[i] = -1;
  if (i < 16) accum[i] = 0.0f;
  if (i < D_) zerorow[i] = 0.0f;
}
__global__ void k_scatter(const int* __restrict__ trg_idx, int* owner) {
  owner[trg_idx[threadIdx.x]] = (int)threadIdx.x;
}

/* ------- per-row: norm->f16/f32, logits, softmax, ent, CE, mp ------- */
__global__ __launch_bounds__(256) void k_rows(
    const float* __restrict__ trg_feat, const float* __restrict__ W,
    const float* __restrict__ bias, const float* __restrict__ Waad,
    const float* __restrict__ baad, _Float16* __restrict__ fn16,
    float* __restrict__ fn32, float* __restrict__ paad,
    float* __restrict__ accum) {
  __shared__ float smL[C_ * 256];
  __shared__ float smA[C_ * 256];
  __shared__ float smS[256];
  const int row = blockIdx.x, t = threadIdx.x;
  const float* x = trg_feat + (size_t)row * D_;
  float x0 = x[t], x1 = x[t + 256];
  smS[t] = x0 * x0 + x1 * x1;
  #pragma unroll
  for (int c = 0; c < C_; ++c) {
    smL[c * 256 + t] = x0 * W[t * C_ + c]    + x1 * W[(t + 256) * C_ + c];
    smA[c * 256 + t] = x0 * Waad[t * C_ + c] + x1 * Waad[(t + 256) * C_ + c];
  }
  __syncthreads();
  for (int s = 128; s > 0; s >>= 1) {
    if (t < s) {
      smS[t] += smS[t + s];
      #pragma unroll
      for (int c = 0; c < C_; ++c) {
        smL[c * 256 + t] += smL[c * 256 + t + s];
        smA[c * 256 + t] += smA[c * 256 + t + s];
      }
    }
    __syncthreads();
  }
  float rn = rsqrtf(smS[0]);
  float f0 = x0 * rn, f1 = x1 * rn;
  fn16[(size_t)row * D_ + t]       = (_Float16)f0;
  fn16[(size_t)row * D_ + t + 256] = (_Float16)f1;
  fn32[(size_t)row * D_ + t]       = f0;
  fn32[(size_t)row * D_ + t + 256] = f1;
  if (t == 0) {
    float lg[C_], la[C_], p[C_], pa[C_];
    float mx = -3.0e38f, mxa = -3.0e38f;
    #pragma unroll
    for (int c = 0; c < C_; ++c) {
      lg[c] = smL[c * 256] + bias[c];
      la[c] = smA[c * 256] + baad[c];
      mx = fmaxf(mx, lg[c]); mxa = fmaxf(mxa, la[c]);
    }
    float se = 0.f, sea = 0.f;
    #pragma unroll
    for (int c = 0; c < C_; ++c) {
      p[c]  = expf(lg[c] - mx);  se  += p[c];
      pa[c] = expf(la[c] - mxa); sea += pa[c];
    }
    float ent = 0.f, pmax = 0.f, pamax = 0.f;
    #pragma unroll
    for (int c = 0; c < C_; ++c) {
      p[c] /= se; pa[c] /= sea;
      ent -= p[c] * logf(p[c] + 1e-5f);
      pmax = fmaxf(pmax, p[c]); pamax = fmaxf(pamax, pa[c]);
      paad[row * C_ + c] = pa[c];
    }
    bool pick1 = pmax > pamax;
    int pseudo = 0; float best = -1.f;
    #pragma unroll
    for (int c = 0; c < C_; ++c) {
      float ch = pick1 ? p[c] : pa[c];
      if (ch > best) { best = ch; pseudo = c; }
    }
    float lse = mx + logf(se), ce = 0.f;
    #pragma unroll
    for (int c = 0; c < C_; ++c) {
      float tgt = (c == pseudo ? 0.9f : 0.0f) + 0.01f; /* EPS_LS=0.1, C=10 */
      ce -= tgt * (lg[c] - lse);
    }
    atomicAdd(&accum[0], ent);
    atomicAdd(&accum[1], ce);
    #pragma unroll
    for (int c = 0; c < C_; ++c) atomicAdd(&accum[3 + c], p[c]);
  }
}

/* --- dist = fn @ fb.T fused with streaming per-row top-6 (f16 WMMA) ---
 * A fragments (this wave's two 16-row tiles, full K) live in 256 VGPRs for
 * the whole kernel; the inner loop is pure global_load->cvt->wmma. LDS is
 * only a 16KB wave-private bounce buffer for the top-k ownership flip. */
__global__ __launch_bounds__(256, 1) void k_dist_topk(
    const float* __restrict__ fea_bank, const _Float16* __restrict__ fn16,
    const float* __restrict__ fn32, const float* __restrict__ zerorow,
    const int* __restrict__ owner, float* __restrict__ candv,
    int* __restrict__ candi) {
  __shared__ float scratch[8 * 512]; /* 8 waves * (2 tiles * 256 f32) */
  const int t = threadIdx.x;
  const int w = t >> 5, lane = t & 31;
  const int g = lane >> 4, nn = lane & 15;
  const int colbase = blockIdx.x * CHUNK;

  /* one-time: load this wave's A fragments (rows 32w..32w+31, all K) */
  v16h a0[D_ / 32], a1[D_ / 32];
  {
    const _Float16* r0 = fn16 + (size_t)(32 * w + nn) * D_ + 16 * g;
    const _Float16* r1 = fn16 + (size_t)(32 * w + 16 + nn) * D_ + 16 * g;
    #pragma unroll
    for (int kk = 0; kk < D_ / 32; ++kk) {
      *(uint4*)&a0[kk]         = *(const uint4*)(r0 + kk * 32);
      *(((uint4*)&a0[kk]) + 1) = *(const uint4*)(r0 + kk * 32 + 8);
      *(uint4*)&a1[kk]         = *(const uint4*)(r1 + kk * 32);
      *(((uint4*)&a1[kk]) + 1) = *(const uint4*)(r1 + kk * 32 + 8);
    }
  }

  float tv[6]; int ti[6];
  #pragma unroll
  for (int s = 0; s < 6; ++s) { tv[s] = -3.0e38f; ti[s] = 0; }

  float* ws0 = scratch + w * 512; /* wave-private */

  for (int ct = 0; ct < CHUNK / 16; ++ct) {
    const int col0 = colbase + ct * 16;
    const int j = col0 + nn;
    /* one-time per-tile source select (no divergence in the K-loop) */
    int ow = -2;
    if (j < N_) ow = owner[j];
    const float* bsrc = (ow == -2) ? zerorow
                      : (ow >= 0)  ? (fn32 + (size_t)ow * D_)
                                   : (fea_bank + (size_t)j * D_);
    bsrc += 16 * g;

    v8f acc0 = {}; v8f acc1 = {};
    #pragma unroll
    for (int kk = 0; kk < D_ / 32; ++kk) {
      const int k0 = kk * 32;
      float4 f0 = *(const float4*)(bsrc + k0);
      float4 f1 = *(const float4*)(bsrc + k0 + 4);
      float4 f2 = *(const float4*)(bsrc + k0 + 8);
      float4 f3 = *(const float4*)(bsrc + k0 + 12);
      v16h bfrag;
      bfrag[0]=(_Float16)f0.x;  bfrag[1]=(_Float16)f0.y;
      bfrag[2]=(_Float16)f0.z;  bfrag[3]=(_Float16)f0.w;
      bfrag[4]=(_Float16)f1.x;  bfrag[5]=(_Float16)f1.y;
      bfrag[6]=(_Float16)f1.z;  bfrag[7]=(_Float16)f1.w;
      bfrag[8]=(_Float16)f2.x;  bfrag[9]=(_Float16)f2.y;
      bfrag[10]=(_Float16)f2.z; bfrag[11]=(_Float16)f2.w;
      bfrag[12]=(_Float16)f3.x; bfrag[13]=(_Float16)f3.y;
      bfrag[14]=(_Float16)f3.z; bfrag[15]=(_Float16)f3.w;
      acc0 = __builtin_amdgcn_wmma_f32_16x16x32_f16(false, a0[kk], false, bfrag,
                                                    (short)0, acc0, false, false);
      acc1 = __builtin_amdgcn_wmma_f32_16x16x32_f16(false, a1[kk], false, bfrag,
                                                    (short)0, acc1, false, false);
    }
    /* bounce tiles through wave-private LDS so each lane owns one row */
    *(float4*)(ws0 + lane * 8)     = make_float4(acc0[0], acc0[1], acc0[2], acc0[3]);
    *(float4*)(ws0 + lane * 8 + 4) = make_float4(acc0[4], acc0[5], acc0[6], acc0[7]);
    *(float4*)(ws0 + 256 + lane * 8)     = make_float4(acc1[0], acc1[1], acc1[2], acc1[3]);
    *(float4*)(ws0 + 256 + lane * 8 + 4) = make_float4(acc1[4], acc1[5], acc1[6], acc1[7]);
    __asm__ volatile("s_wait_dscnt 0" ::: "memory"); /* LDS in-order per wave */
    const int tsel = lane >> 4, mm = lane & 15;
    const float* rb = ws0 + tsel * 256 + 128 * (mm >> 3) + (mm & 7);
    #pragma unroll
    for (int n = 0; n < 16; ++n) {
      float v = rb[n * 8];
      int jj = col0 + n;
      if (jj < N_ && v > tv[5]) TOPK_INSERT(tv, ti, v, jj);
    }
  }
  const int row = 32 * w + lane;
  size_t cb = ((size_t)row * NCHUNK + blockIdx.x) * 6;
  #pragma unroll
  for (int s = 0; s < 6; ++s) { candv[cb + s] = tv[s]; candi[cb + s] = ti[s]; }
}

/* ------- merge candidates -> top-6, drop self, KL(loss_aad) ------- */
__global__ __launch_bounds__(256) void k_merge(
    const float* __restrict__ candv, const int* __restrict__ candi,
    const int* __restrict__ owner, const float* __restrict__ paad,
    const float* __restrict__ score_bank, float* __restrict__ accum) {
  const int row = threadIdx.x;
  float tv[6]; int ti[6];
  #pragma unroll
  for (int s = 0; s < 6; ++s) { tv[s] = -3.0e38f; ti[s] = 0; }
  const float* cv = candv + (size_t)row * NCHUNK * 6;
  const int*   ci = candi + (size_t)row * NCHUNK * 6;
  for (int e = 0; e < NCHUNK * 6; ++e) {
    float v = cv[e];
    if (v > tv[5]) { int jj = ci[e]; TOPK_INSERT(tv, ti, v, jj); }
  }
  const float* prow = paad + row * C_;
  float kl = 0.f;
  #pragma unroll
  for (int s = 1; s < 6; ++s) { /* drop best (self column), keep next 5 */
    int nb = ti[s];
    int ow = owner[nb];
    const float* srow = (ow >= 0) ? (paad + (size_t)ow * C_)
                                  : (score_bank + (size_t)nb * C_);
    #pragma unroll
    for (int c = 0; c < C_; ++c) {
      float sc = srow[c];
      kl += sc * (logf(sc) - prow[c]);
    }
  }
  atomicAdd(&accum[2], kl);
}

/* ---------------- final scalar combine ---------------- */
__global__ void k_final(const float* __restrict__ accum, float* __restrict__ out) {
  float ent = accum[0] / (float)B_;
  float ce  = accum[1] / (float)B_;
  float kl  = accum[2] / (float)B_;
  float im = 0.f;
  #pragma unroll
  for (int c = 0; c < C_; ++c) {
    float mp = accum[3 + c] / (float)B_;
    im += -mp * logf(mp + 1e-5f);
  }
  out[0] = 1.0f * ent - 1.0f * im + 1.0f * kl + 0.1f * ce;
}

extern "C" void kernel_launch(void* const* d_in, const int* in_sizes, int n_in,
                              void* d_out, int out_size, void* d_ws, size_t ws_size,
                              hipStream_t stream) {
  const float* trg_feat   = (const float*)d_in[0];
  const float* W          = (const float*)d_in[1];
  const float* bias       = (const float*)d_in[2];
  const float* Waad       = (const float*)d_in[3];
  const float* baad       = (const float*)d_in[4];
  const float* fea_bank   = (const float*)d_in[5];
  const float* score_bank = (const float*)d_in[6];
  const int*   trg_idx    = (const int*)d_in[7];
  (void)in_sizes; (void)n_in; (void)out_size; (void)ws_size;

  char* ws = (char*)d_ws;
  _Float16* fn16 = (_Float16*)(ws + OFF_FN16);
  float*    fn32 = (float*)(ws + OFF_FN32);
  float*    zerorow = (float*)(ws + OFF_ZERO);
  float*    paad = (float*)(ws + OFF_PAAD);
  int*      owner = (int*)(ws + OFF_OWNER);
  float*    accum = (float*)(ws + OFF_ACCUM);
  float*    candv = (float*)(ws + OFF_CANDV);
  int*      candi = (int*)(ws + OFF_CANDI);
  float*    out = (float*)d_out;

  k_init<<<(N_ + 255) / 256, 256, 0, stream>>>(owner, accum, zerorow);
  k_scatter<<<1, 256, 0, stream>>>(trg_idx, owner);
  k_rows<<<B_, 256, 0, stream>>>(trg_feat, W, bias, Waad, baad, fn16, fn32, paad, accum);
  k_dist_topk<<<NCHUNK, 256, 0, stream>>>(fea_bank, fn16, fn32, zerorow,
                                          owner, candv, candi);
  k_merge<<<1, 256, 0, stream>>>(candv, candi, owner, paad, score_bank, accum);
  k_final<<<1, 1, 0, stream>>>(accum, out);
}
